// Controller_87565793231276
// MI455X (gfx1250) — compile-verified, matrix operand order checked by reading, not run on hardware
//
#include <hip/hip_runtime.h>

// ---------------- problem constants ----------------
#define PB      512      // batch
#define PH      1024     // hidden per layer
#define PL      4        // layers
#define NGATE   4
#define INDIM   2560     // 512 + 1024 + 1024
#define PX      512
#define PVT     512
#define PET     256
#define LH      4096     // PL * PH

typedef __attribute__((ext_vector_type(16))) __bf16 v16bf;
typedef __attribute__((ext_vector_type(8)))  float  v8f;
typedef int v4i __attribute__((vector_size(16)));

union Frag { uint4 u[2]; v16bf v; };

#define AS1 __attribute__((address_space(1)))
#define AS3 __attribute__((address_space(3)))

#if __has_builtin(__builtin_amdgcn_global_load_async_to_lds_b128) && \
    __has_builtin(__builtin_amdgcn_s_wait_asynccnt)
#define HAS_ASYNC_LDS 1
#else
#define HAS_ASYNC_LDS 0
#endif

// copy 16B global -> LDS (async on CDNA5; sync fallback otherwise)
__device__ __forceinline__ void cp16(const void* g, void* l) {
#if HAS_ASYNC_LDS
    __builtin_amdgcn_global_load_async_to_lds_b128(
        (AS1 v4i*)g, (AS3 v4i*)l, 0, 0);
#else
    *(uint4*)l = *(const uint4*)g;
#endif
}
__device__ __forceinline__ void async_wait() {
#if HAS_ASYNC_LDS
    __builtin_amdgcn_s_wait_asynccnt(0);
#endif
}

__device__ __forceinline__ unsigned short f2bf(float f) {
    unsigned u = __builtin_bit_cast(unsigned, f);
    u += 0x7FFFu + ((u >> 16) & 1u);           // round-to-nearest-even
    return (unsigned short)(u >> 16);
}
__device__ __forceinline__ unsigned pk2(float a, float b) {
    return (unsigned)f2bf(a) | ((unsigned)f2bf(b) << 16);
}
__device__ __forceinline__ float sigmoidf_(float x) {
    return 1.0f / (1.0f + __expf(-x));
}

// ---------------- one-shot fp32 -> bf16 weight conversion ----------------
__global__ void cvt_kernel(const float* __restrict__ src,
                           unsigned short* __restrict__ dst, size_t n) {
    size_t i = ((size_t)blockIdx.x * blockDim.x + threadIdx.x) * 8;
    if (i >= n) return;
    float4 a = *(const float4*)(src + i);
    float4 b = *(const float4*)(src + i + 4);
    *(uint4*)(dst + i) = make_uint4(pk2(a.x, a.y), pk2(a.z, a.w),
                                    pk2(b.x, b.y), pk2(b.z, b.w));
}

// ---------------- prep: build bf16 z buffers ----------------
// zbuf: [PL][PB][INDIM] bf16; cols [0,512)=x, [512,1536)=h_prev[l], [1536,2560)=prev-layer h
__global__ void prep_kernel(const float* __restrict__ x,
                            const float* __restrict__ h_prev,
                            unsigned short* __restrict__ zbuf) {
    size_t idx = (size_t)blockIdx.x * blockDim.x + threadIdx.x;
    const size_t total = (size_t)PL * PB * INDIM;
    if (idx >= total) return;
    int l = (int)(idx / ((size_t)PB * INDIM));
    int rem = (int)(idx % ((size_t)PB * INDIM));
    int b = rem / INDIM;
    int d = rem % INDIM;
    float v;
    if (d < PX)            v = x[(size_t)b * PX + d];
    else if (d < PX + PH)  v = h_prev[((size_t)l * PB + b) * PH + (d - PX)];
    else                   v = 0.0f;   // layer0 zeros; l>0 overwritten by layer l-1 epilogue
    zbuf[idx] = f2bf(v);
}

// ---------------- fused LSTM layer ----------------
// grid (PB/64, PH/64), block 256 (8 waves). Per wave: 32(b) x 16(h) x 4 gates.
// Double-buffered LDS, async global->LDS copies.
#define ROWB    80                    // LDS row pitch bytes (32 bf16 + pad)
#define ATILE_B (64 * ROWB)           // 5120
#define BTILE_B (64 * ROWB)           // per gate
#define BUFB    (ATILE_B + NGATE * BTILE_B)   // 25600 per buffer

__global__ __launch_bounds__(256)
void lstm_layer_kernel(const unsigned short* __restrict__ zl,    // [PB][INDIM] bf16
                       const unsigned short* __restrict__ Wgb,   // [PL][4][PH][INDIM] bf16
                       const float* __restrict__ bg,             // [PL][4][PH]
                       const float* __restrict__ c_prev,         // [PL][PB][PH]
                       unsigned short* __restrict__ hidden,      // [PB][LH] bf16
                       unsigned short* __restrict__ znext,       // next layer zbuf or null
                       int l) {
    __shared__ unsigned char smem[2 * BUFB];

    const int t  = threadIdx.x;
    const int b0 = blockIdx.x * 64;
    const int h0 = blockIdx.y * 64;

    const int lane = t & 31;
    const int wave = t >> 5;
    const int wb = wave >> 2;            // 0..1 : 32-row sub-tile
    const int wh = wave & 3;             // 0..3 : 16-col sub-tile
    const int m = lane & 15;
    const int half = lane >> 4;

    // copy coordinates: each thread moves 16B of A and 4x16B of B per k-step
    const int crow = t >> 2;             // 0..63
    const int ckc  = t & 3;              // 0..3 : 16B chunk within 64B row
    const unsigned short* asrc_row = zl + (size_t)(b0 + crow) * INDIM + ckc * 8;
    const unsigned short* bsrc_row[NGATE];
#pragma unroll
    for (int g = 0; g < NGATE; ++g)
        bsrc_row[g] = Wgb + ((size_t)(l * NGATE + g) * PH + (h0 + crow)) * INDIM + ckc * 8;
    const int ldst = crow * ROWB + ckc * 16;

    v8f acc0[NGATE], acc1[NGATE];
#pragma unroll
    for (int g = 0; g < NGATE; ++g)
#pragma unroll
        for (int j = 0; j < 8; ++j) { acc0[g][j] = 0.0f; acc1[g][j] = 0.0f; }

    // prologue: issue tile k=0 into buffer 0
    {
        unsigned char* base = smem;
        cp16(asrc_row, base + ldst);
#pragma unroll
        for (int g = 0; g < NGATE; ++g)
            cp16(bsrc_row[g], base + ATILE_B + g * BTILE_B + ldst);
    }

    for (int k0 = 0; k0 < INDIM; k0 += 32) {
        const int cur = (k0 >> 5) & 1;
        async_wait();            // own async copies have landed in LDS
        __syncthreads();         // everyone's copies landed; prev-iter reads done

        if (k0 + 32 < INDIM) {   // prefetch next tile into the other buffer
            unsigned char* base = smem + (cur ^ 1) * BUFB;
            cp16(asrc_row + k0 + 32, base + ldst);
#pragma unroll
            for (int g = 0; g < NGATE; ++g)
                cp16(bsrc_row[g] + k0 + 32, base + ATILE_B + g * BTILE_B + ldst);
        }

        const unsigned char* sA = smem + cur * BUFB;
        const unsigned char* sB = sA + ATILE_B;

        Frag a0, a1;
        const unsigned char* ab0 = sA + (wb * 32 + m) * ROWB;
        a0.u[0] = *(const uint4*)(ab0 + half * 16);
        a0.u[1] = *(const uint4*)(ab0 + 32 + half * 16);
        const unsigned char* ab1 = ab0 + 16 * ROWB;
        a1.u[0] = *(const uint4*)(ab1 + half * 16);
        a1.u[1] = *(const uint4*)(ab1 + 32 + half * 16);

#pragma unroll
        for (int g = 0; g < NGATE; ++g) {
            Frag bf;
            const unsigned char* bb = sB + g * BTILE_B + (wh * 16 + m) * ROWB;
            bf.u[0] = *(const uint4*)(bb + half * 16);
            bf.u[1] = *(const uint4*)(bb + 32 + half * 16);
            acc0[g] = __builtin_amdgcn_wmma_f32_16x16x32_bf16(
                false, a0.v, false, bf.v, (short)0, acc0[g], false, false);
            acc1[g] = __builtin_amdgcn_wmma_f32_16x16x32_bf16(
                false, a1.v, false, bf.v, (short)0, acc1[g], false, false);
        }
    }

    // fused LSTM epilogue (all 4 gate values for (b,h) live in the same lane/slot)
    const int h = h0 + wh * 16 + m;
    const float bi  = bg[((size_t)l * NGATE + 0) * PH + h];
    const float bff = bg[((size_t)l * NGATE + 1) * PH + h];
    const float bo  = bg[((size_t)l * NGATE + 2) * PH + h];
    const float bs  = bg[((size_t)l * NGATE + 3) * PH + h];
#pragma unroll
    for (int r = 0; r < 2; ++r) {
        const v8f* ar = r ? acc1 : acc0;
#pragma unroll
        for (int j = 0; j < 8; ++j) {
            const int b = b0 + wb * 32 + r * 16 + j + 8 * half;
            float gi = sigmoidf_(ar[0][j] + bi);
            float gf = sigmoidf_(ar[1][j] + bff);
            float go = sigmoidf_(ar[2][j] + bo);
            float gs = tanhf(ar[3][j] + bs);
            float c  = gf * c_prev[((size_t)l * PB + b) * PH + h] + gi * gs;
            float hn = go * tanhf(c);
            unsigned short hb = f2bf(hn);
            hidden[(size_t)b * LH + l * PH + h] = hb;
            if (znext) znext[(size_t)b * INDIM + (PX + PH) + h] = hb;
        }
    }
}

// ---------------- final projections ----------------
// grid (PB/64, 12), block 256. Per wave: 32(b) x 16(n). N = [W_y rows | W_E rows].
#define PBUFB (ATILE_B + BTILE_B)     // 10240 per buffer

__global__ __launch_bounds__(256)
void proj_kernel(const unsigned short* __restrict__ hidden,   // [PB][LH] bf16
                 const unsigned short* __restrict__ Wyb,      // [PVT][LH] bf16
                 const float* __restrict__ by,
                 const unsigned short* __restrict__ WEb,      // [PET][LH] bf16
                 const float* __restrict__ bE,
                 float* __restrict__ out) {
    __shared__ unsigned char smem[2 * PBUFB];

    const int t  = threadIdx.x;
    const int b0 = blockIdx.x * 64;
    const int n0 = blockIdx.y * 64;
    const bool isY = (n0 < PVT);
    const unsigned short* Wmat = isY ? Wyb : WEb;
    const int nbase = isY ? n0 : (n0 - PVT);

    const int lane = t & 31;
    const int wave = t >> 5;
    const int wb = wave >> 2, wh = wave & 3;
    const int m = lane & 15, half = lane >> 4;

    const int crow = t >> 2, ckc = t & 3;
    const unsigned short* asrc_row = hidden + (size_t)(b0 + crow) * LH + ckc * 8;
    const unsigned short* bsrc_row = Wmat + (size_t)(nbase + crow) * LH + ckc * 8;
    const int ldst = crow * ROWB + ckc * 16;

    v8f acc0, acc1;
#pragma unroll
    for (int j = 0; j < 8; ++j) { acc0[j] = 0.0f; acc1[j] = 0.0f; }

    cp16(asrc_row, smem + ldst);
    cp16(bsrc_row, smem + ATILE_B + ldst);

    for (int k0 = 0; k0 < LH; k0 += 32) {
        const int cur = (k0 >> 5) & 1;
        async_wait();
        __syncthreads();
        if (k0 + 32 < LH) {
            unsigned char* base = smem + (cur ^ 1) * PBUFB;
            cp16(asrc_row + k0 + 32, base + ldst);
            cp16(bsrc_row + k0 + 32, base + ATILE_B + ldst);
        }
        const unsigned char* sA = smem + cur * PBUFB;
        const unsigned char* sB = sA + ATILE_B;

        Frag a0, a1, bf;
        const unsigned char* ab0 = sA + (wb * 32 + m) * ROWB;
        a0.u[0] = *(const uint4*)(ab0 + half * 16);
        a0.u[1] = *(const uint4*)(ab0 + 32 + half * 16);
        const unsigned char* ab1 = ab0 + 16 * ROWB;
        a1.u[0] = *(const uint4*)(ab1 + half * 16);
        a1.u[1] = *(const uint4*)(ab1 + 32 + half * 16);
        const unsigned char* bb = sB + (wh * 16 + m) * ROWB;
        bf.u[0] = *(const uint4*)(bb + half * 16);
        bf.u[1] = *(const uint4*)(bb + 32 + half * 16);

        acc0 = __builtin_amdgcn_wmma_f32_16x16x32_bf16(
            false, a0.v, false, bf.v, (short)0, acc0, false, false);
        acc1 = __builtin_amdgcn_wmma_f32_16x16x32_bf16(
            false, a1.v, false, bf.v, (short)0, acc1, false, false);
    }

    const int n = n0 + wh * 16 + m;
    const float bias = isY ? by[n] : bE[n - PVT];
#pragma unroll
    for (int r = 0; r < 2; ++r) {
        const v8f& ar = r ? acc1 : acc0;
#pragma unroll
        for (int j = 0; j < 8; ++j) {
            const int b = b0 + wb * 32 + r * 16 + j + 8 * half;
            float v = ar[j] + bias;
            if (isY) out[(size_t)b * PVT + n] = v;
            else     out[(size_t)PB * PVT + (size_t)b * PET + (n - PVT)] = v;
        }
    }
}

// ---------------- launch ----------------
extern "C" void kernel_launch(void* const* d_in, const int* in_sizes, int n_in,
                              void* d_out, int out_size, void* d_ws, size_t ws_size,
                              hipStream_t stream) {
    const float* x      = (const float*)d_in[0];
    const float* h_prev = (const float*)d_in[1];
    const float* c_prev = (const float*)d_in[2];
    const float* Wg     = (const float*)d_in[3];
    const float* bg     = (const float*)d_in[4];
    const float* Wy     = (const float*)d_in[5];
    const float* by     = (const float*)d_in[6];
    const float* WE     = (const float*)d_in[7];
    const float* bE     = (const float*)d_in[8];
    float* out = (float*)d_out;

    const size_t nWg = (size_t)PL * NGATE * PH * INDIM;   // 41,943,040
    const size_t nWy = (size_t)PVT * LH;                  //  2,097,152
    const size_t nWE = (size_t)PET * LH;                  //  1,048,576
    unsigned short* Wgb    = (unsigned short*)d_ws;
    unsigned short* Wyb    = Wgb + nWg;
    unsigned short* WEb    = Wyb + nWy;
    unsigned short* zbuf   = WEb + nWE;                   // PL*PB*INDIM
    unsigned short* hidden = zbuf + (size_t)PL * PB * INDIM;

    cvt_kernel<<<(int)(nWg / 8 / 256), 256, 0, stream>>>(Wg, Wgb, nWg);
    cvt_kernel<<<(int)(nWy / 8 / 256), 256, 0, stream>>>(Wy, Wyb, nWy);
    cvt_kernel<<<(int)(nWE / 8 / 256), 256, 0, stream>>>(WE, WEb, nWE);
    {
        const size_t total = (size_t)PL * PB * INDIM;
        prep_kernel<<<(int)((total + 255) / 256), 256, 0, stream>>>(x, h_prev, zbuf);
    }
    for (int l = 0; l < PL; ++l) {
        unsigned short* zl = zbuf + (size_t)l * PB * INDIM;
        unsigned short* zn = (l < PL - 1) ? (zbuf + (size_t)(l + 1) * PB * INDIM) : nullptr;
        lstm_layer_kernel<<<dim3(PB / 64, PH / 64), 256, 0, stream>>>(
            zl, Wgb, bg, c_prev, hidden, zn, l);
    }
    proj_kernel<<<dim3(PB / 64, (PVT + PET) / 64), 256, 0, stream>>>(
        hidden, Wyb, by, WEb, bE, out);
}
